// EGNN_62612033241377
// MI455X (gfx1250) — matrix operand order repaired and checked by reading, not compiled, not run on hardware
//
#include <hip/hip_runtime.h>
#include <math.h>

#define NN 1024
#define FF 64
#define DD 2
#define PI_F 3.14159265358979323846f

typedef __attribute__((ext_vector_type(16))) _Float16 v16h;
typedef __attribute__((ext_vector_type(8)))  _Float16 v8h;
typedef __attribute__((ext_vector_type(8)))  float    v8f;
typedef __attribute__((ext_vector_type(4)))  float    v4f;

__device__ __forceinline__ v8f wmma32(v16h a, v16h b, v8f c) {
  return __builtin_amdgcn_wmma_f32_16x16x32_f16(false, a, false, b, (short)0, c,
                                                false, false);
}

__device__ __forceinline__ float silu_f(float x) {
  return x / (1.0f + __expf(-x));
}

// A-fragment K index for element idx (0..15): half=0 -> K 0..7,16..23 ; half=1 -> K 8..15,24..31
__device__ __forceinline__ int afrag_k(int k0, int half, int idx) {
  return k0 + half * 8 + (idx & 7) + ((idx >> 3) << 4);
}

// B-fragment (KxN, row-major f32 weights, ldn columns): lanes 0-15 = col, halves split K
__device__ __forceinline__ v16h load_b_w(const float* __restrict__ W, int ldn,
                                         int col, int half, int k0) {
  v16h b;
  int kb = k0 + half * 8;
#pragma unroll
  for (int t = 0; t < 16; ++t) b[t] = (_Float16)W[(kb + (t & 7) + ((t >> 3) << 4)) * ldn + col];
  return b;
}

// rebuild a f16 A-fragment from a contiguous f32 row in LDS using 16B reads
__device__ __forceinline__ v16h afrag_from_row(const v4f* __restrict__ srow,
                                               int half, int kc) {
  int g = kc * 8 + half * 2;  // (kc*32 + half*8) / 4
  v4f q0 = srow[g], q1 = srow[g + 1];      // K kb .. kb+7
  v4f q2 = srow[g + 4], q3 = srow[g + 5];  // K kb+16 .. kb+23
  v16h a;
#pragma unroll
  for (int t = 0; t < 4; ++t) {
    a[t]      = (_Float16)q0[t];
    a[4 + t]  = (_Float16)q1[t];
    a[8 + t]  = (_Float16)q2[t];
    a[12 + t] = (_Float16)q3[t];
  }
  return a;
}

// ---------------------------------------------------------------- init
__global__ void init_kernel(const float* __restrict__ x, float* __restrict__ h0,
                            float* __restrict__ x0) {
  int idx = blockIdx.x * blockDim.x + threadIdx.x;
  if (idx < NN * FF) h0[idx] = 1.0f;
  if (idx < NN * 3) x0[idx] = x[idx];
}

// per-depth prep: zero m accumulator, seed xnext with current x
__global__ void prep_edge_kernel(const float* __restrict__ xc,
                                 float* __restrict__ xnext,
                                 float* __restrict__ mb) {
  int idx = blockIdx.x * blockDim.x + threadIdx.x;
  if (idx < NN * FF) mb[idx] = 0.0f;
  if (idx < NN * 3) xnext[idx] = xc[idx];
}

// ----------------------------- A = h@W1a + b1 (f32) ; B = h@W1b (f16)
__global__ void __launch_bounds__(256)
gemm_ab_kernel(const float* __restrict__ h, const float* __restrict__ ew1,
               const float* __restrict__ eb1, float* __restrict__ Ab,
               _Float16* __restrict__ Bh) {
  int wave = threadIdx.x >> 5;
  int lane = threadIdx.x & 31;
  int half = lane >> 4, lc = lane & 15;
  int wg = blockIdx.x * 8 + wave;       // 512 wave-tiles total
  int which = wg & 1;                   // 0: A-part, 1: B-part
  int nt = (wg >> 1) & 3;
  int mt = wg >> 3;
  const float* W = ew1 + (which ? FF * FF : 0);  // rows 0..63 or 64..127
  v8f acc = {};
#pragma unroll
  for (int kc = 0; kc < 2; ++kc) {
    int k0 = kc * 32;
    v16h a;
#pragma unroll
    for (int idx = 0; idx < 16; ++idx)
      a[idx] = (_Float16)h[(mt * 16 + lc) * FF + afrag_k(k0, half, idx)];
    v16h b = load_b_w(W, FF, nt * 16 + lc, half, k0);
    acc = wmma32(a, b, acc);
  }
  if (which) {
#pragma unroll
    for (int v = 0; v < 8; ++v)
      Bh[(mt * 16 + v + half * 8) * FF + nt * 16 + lc] = (_Float16)acc[v];
  } else {
    float bias = eb1[nt * 16 + lc];
#pragma unroll
    for (int v = 0; v < 8; ++v)
      Ab[(mt * 16 + v + half * 8) * FF + nt * 16 + lc] = acc[v] + bias;
  }
}

// --------------------------------------------------------- fused edge pass
// two waves per row i (each covers half the j range) -> 2 waves/SIMD residency
__global__ void __launch_bounds__(256)
edge_kernel(const float* __restrict__ xc, const float* __restrict__ Ab,
            const _Float16* __restrict__ Bh, const float* __restrict__ w1r,
            const float* __restrict__ eb2, const float* __restrict__ ew2,
            const float* __restrict__ cw1, const float* __restrict__ cb1,
            const float* __restrict__ cw2, const float* __restrict__ cb2,
            float* __restrict__ mb, float* __restrict__ xnext) {
  __shared__ float stage[8][16][FF];   // per-wave 16x64 staging tile
  __shared__ float cw2s[FF];
  int wave = threadIdx.x >> 5;
  int lane = threadIdx.x & 31;
  int half = lane >> 4, lc = lane & 15;
  int gw = blockIdx.x * 8 + wave;
  int i = gw >> 1;           // node row
  int jh = gw & 1;           // which half of the j range
  int jbeg = jh * (NN / 2), jend = jbeg + NN / 2;

  if (threadIdx.x < FF) cw2s[threadIdx.x] = cw2[threadIdx.x];
  __syncthreads();

  // preload weight B-fragments (held in VGPRs across the whole j loop)
  v16h w2f[2][4], c1f[2][4];
#pragma unroll
  for (int kc = 0; kc < 2; ++kc)
#pragma unroll
    for (int nt = 0; nt < 4; ++nt) {
      w2f[kc][nt] = load_b_w(ew2, FF, nt * 16 + lc, half, kc * 32);
      c1f[kc][nt] = load_b_w(cw1, FF, nt * 16 + lc, half, kc * 32);
    }
  float b2f[4], cb1f[4];
#pragma unroll
  for (int nt = 0; nt < 4; ++nt) {
    b2f[nt]  = eb2[nt * 16 + lc];
    cb1f[nt] = cb1[nt * 16 + lc];
  }
  // loop-invariant per-lane operands of edge layer 1
  float ai[2][16], wr[2][16];
#pragma unroll
  for (int kc = 0; kc < 2; ++kc)
#pragma unroll
    for (int idx = 0; idx < 16; ++idx) {
      int f = afrag_k(kc * 32, half, idx);
      ai[kc][idx] = Ab[i * FF + f];
      wr[kc][idx] = w1r[f];
    }
  float cb2v = cb2[0];
  float xi0 = xc[i * 3 + 0], xi1 = xc[i * 3 + 1], xi2 = xc[i * 3 + 2];
  float macc[4] = {0.f, 0.f, 0.f, 0.f};
  float dxa0 = 0.f, dxa1 = 0.f, dxa2 = 0.f;

  for (int j0 = jbeg; j0 < jend; j0 += 16) {
    int je = j0 + lc;  // edge index within tile == A-frag row
    if (j0 + 16 < jend) __builtin_prefetch(&Bh[(je + 16) * FF], 0, 3);
    float d0 = xi0 - xc[je * 3 + 0];
    float d1 = xi1 - xc[je * 3 + 1];
    float d2 = xi2 - xc[je * 3 + 2];
    float s0 = __sinf(PI_F * d0), s1 = __sinf(PI_F * d1), s2 = __sinf(PI_F * d2);
    float r = sqrtf(s0 * s0 + s1 * s1 + s2 * s2);
    if (je == i) r = 0.0f;  // (1-eye) factor

    // edge layer1: silu(A_i + B_j + r*w1r) built directly in A-frag layout.
    const v8h* brow = (const v8h*)(Bh + je * FF);
    v16h af[2];
#pragma unroll
    for (int kc = 0; kc < 2; ++kc) {
      v8h blo = brow[kc * 4 + half];      // K = k0 + half*8 + 0..7
      v8h bhi = brow[kc * 4 + half + 2];  // K = k0 + half*8 + 16..23
#pragma unroll
      for (int idx = 0; idx < 8; ++idx) {
        float pre = ai[kc][idx] + (float)blo[idx] + r * wr[kc][idx];
        af[kc][idx] = (_Float16)silu_f(pre);
      }
#pragma unroll
      for (int idx = 8; idx < 16; ++idx) {
        float pre = ai[kc][idx] + (float)bhi[idx - 8] + r * wr[kc][idx];
        af[kc][idx] = (_Float16)silu_f(pre);
      }
    }
    // edge layer2: mij = hid @ W2 + b2 ; accumulate m_i (diagonal excluded)
#pragma unroll
    for (int nt = 0; nt < 4; ++nt) {
      v8f acc = {};
      acc = wmma32(af[0], w2f[0][nt], acc);
      acc = wmma32(af[1], w2f[1][nt], acc);
      int f = nt * 16 + lc;
#pragma unroll
      for (int v = 0; v < 8; ++v) {
        int e = v + half * 8;
        float mv = acc[v] + b2f[nt];
        stage[wave][e][f] = mv;
        if (j0 + e != i) macc[nt] += mv;
      }
    }
    // coord layer1: chid = silu(mij @ cW1 + cb1)  (re-layout via LDS, 16B reads)
    const v4f* srow = (const v4f*)(&stage[wave][lc][0]);
    v16h ma[2];
#pragma unroll
    for (int kc = 0; kc < 2; ++kc) ma[kc] = afrag_from_row(srow, half, kc);
#pragma unroll
    for (int nt = 0; nt < 4; ++nt) {
      v8f acc = {};
      acc = wmma32(ma[0], c1f[0][nt], acc);
      acc = wmma32(ma[1], c1f[1][nt], acc);
      int f = nt * 16 + lc;
#pragma unroll
      for (int v = 0; v < 8; ++v)
        stage[wave][v + half * 8][f] = silu_f(acc[v] + cb1f[nt]);
    }
    // coord layer2: per-edge dot split across the two half-waves, 16B reads
    float wsum = 0.f;
    const v4f* crow = (const v4f*)(&cw2s[0]);
#pragma unroll
    for (int g = 0; g < 8; ++g) {
      v4f q = srow[half * 8 + g];
      v4f c = crow[half * 8 + g];
      wsum += q[0] * c[0] + q[1] * c[1] + q[2] * c[2] + q[3] * c[3];
    }
    wsum += __shfl_xor(wsum, 16, 32);  // lanes 0-15 now hold the full dot
    if (half == 0) {
      wsum += cb2v;
      if (je != i) {
        float m0 = d0 - rintf(d0);           // L = 1
        float m1 = d1 - rintf(d1);
        float m2 = d2 - rintf(d2);
        float rij = sqrtf(m0 * m0 + m1 * m1 + m2 * m2);
        float sw;
        if (rij < 0.25f)      sw = 1.0f;
        else if (rij < 0.5f)  sw = 0.5f * __cosf(PI_F * (rij - 0.25f) * 4.0f) + 0.5f;
        else                  sw = 0.0f;
        float wf = wsum * (1.0f / (float)(NN - 1)) * sw;
        dxa0 += m0 * wf; dxa1 += m1 * wf; dxa2 += m2 * wf;
      }
    }
  }
  // combine half-wave partials per feature; two waves per row -> atomic add
#pragma unroll
  for (int nt = 0; nt < 4; ++nt) {
    float v = macc[nt] + __shfl_xor(macc[nt], 16, 32);
    if (half == 0) atomicAdd(&mb[i * FF + nt * 16 + lc], v);
  }
  // reduce coordinate delta over lanes 0..15
#pragma unroll
  for (int msk = 8; msk >= 1; msk >>= 1) {
    dxa0 += __shfl_xor(dxa0, msk, 32);
    dxa1 += __shfl_xor(dxa1, msk, 32);
    dxa2 += __shfl_xor(dxa2, msk, 32);
  }
  if (lane == 0) {
    atomicAdd(&xnext[i * 3 + 0], dxa0);
    atomicAdd(&xnext[i * 3 + 1], dxa1);
    atomicAdd(&xnext[i * 3 + 2], dxa2);
  }
}

// --------------------------------------------------------------- node MLP
__global__ void __launch_bounds__(256)
node_kernel(const float* __restrict__ hb, const float* __restrict__ mb,
            const float* __restrict__ nw1, const float* __restrict__ nb1,
            const float* __restrict__ nw2, const float* __restrict__ nb2,
            float* __restrict__ hnext) {
  __shared__ float stage[8][16][FF];
  int wave = threadIdx.x >> 5;
  int lane = threadIdx.x & 31;
  int half = lane >> 4, lc = lane & 15;
  int mt = blockIdx.x * 8 + wave;
  int row = mt * 16 + lc;

  v16h w1f[4][4], w2f[2][4];
#pragma unroll
  for (int kc = 0; kc < 4; ++kc)
#pragma unroll
    for (int nt = 0; nt < 4; ++nt)
      w1f[kc][nt] = load_b_w(nw1, FF, nt * 16 + lc, half, kc * 32);
#pragma unroll
  for (int kc = 0; kc < 2; ++kc)
#pragma unroll
    for (int nt = 0; nt < 4; ++nt)
      w2f[kc][nt] = load_b_w(nw2, FF, nt * 16 + lc, half, kc * 32);

  v8f acc[4] = {};
#pragma unroll
  for (int kc = 0; kc < 4; ++kc) {  // K = 128: concat(h, m)
    v16h a;
#pragma unroll
    for (int idx = 0; idx < 16; ++idx) {
      int k = afrag_k(kc * 32, half, idx);
      float v = (k < FF) ? hb[row * FF + k] : mb[row * FF + (k - FF)];
      a[idx] = (_Float16)v;
    }
#pragma unroll
    for (int nt = 0; nt < 4; ++nt) acc[nt] = wmma32(a, w1f[kc][nt], acc[nt]);
  }
#pragma unroll
  for (int nt = 0; nt < 4; ++nt) {
    int f = nt * 16 + lc;
    float b = nb1[f];
#pragma unroll
    for (int v = 0; v < 8; ++v)
      stage[wave][v + half * 8][f] = silu_f(acc[nt][v] + b);
  }
  const v4f* srow = (const v4f*)(&stage[wave][lc][0]);
  v16h ha[2];
#pragma unroll
  for (int kc = 0; kc < 2; ++kc) ha[kc] = afrag_from_row(srow, half, kc);
#pragma unroll
  for (int nt = 0; nt < 4; ++nt) {
    v8f a2 = {};
    a2 = wmma32(ha[0], w2f[0][nt], a2);
    a2 = wmma32(ha[1], w2f[1][nt], a2);
    int f = nt * 16 + lc;
    float b = nb2[f];
#pragma unroll
    for (int v = 0; v < 8; ++v) {
      int rr = mt * 16 + v + half * 8;
      hnext[rr * FF + f] = a2[v] + b + hb[rr * FF + f];  // residual
    }
  }
}

// ------------------------------------------------------------ final proj
__global__ void final_kernel(const float* __restrict__ hb,
                             const float* __restrict__ xb,
                             const float* __restrict__ fw,
                             float* __restrict__ out) {
  int i = blockIdx.x * blockDim.x + threadIdx.x;
  if (i >= NN) return;
  float a0 = 0.f, a1 = 0.f, a2 = 0.f;
#pragma unroll
  for (int f = 0; f < FF; ++f) {
    float hv = hb[i * FF + f];
    a0 += hv * fw[f * 3 + 0];
    a1 += hv * fw[f * 3 + 1];
    a2 += hv * fw[f * 3 + 2];
  }
  out[i * 3 + 0] = a0 + xb[i * 3 + 0];
  out[i * 3 + 1] = a1 + xb[i * 3 + 1];
  out[i * 3 + 2] = a2 + xb[i * 3 + 2];
}

extern "C" void kernel_launch(void* const* d_in, const int* in_sizes, int n_in,
                              void* d_out, int out_size, void* d_ws, size_t ws_size,
                              hipStream_t stream) {
  const float* x   = (const float*)d_in[0];
  const float* ew1 = (const float*)d_in[1];
  const float* eb1 = (const float*)d_in[2];
  const float* ew2 = (const float*)d_in[3];
  const float* eb2 = (const float*)d_in[4];
  const float* cw1 = (const float*)d_in[5];
  const float* cb1 = (const float*)d_in[6];
  const float* cw2 = (const float*)d_in[7];
  const float* cb2 = (const float*)d_in[8];
  const float* nw1 = (const float*)d_in[9];
  const float* nb1 = (const float*)d_in[10];
  const float* nw2 = (const float*)d_in[11];
  const float* nb2 = (const float*)d_in[12];
  const float* fw  = (const float*)d_in[13];
  float* out = (float*)d_out;

  float* ws = (float*)d_ws;
  float* xb[2] = {ws, ws + NN * 3};
  float* hb[2] = {ws + 2 * NN * 3, ws + 2 * NN * 3 + NN * FF};
  float* Ab = ws + 2 * NN * 3 + 2 * NN * FF;
  float* mb = Ab + NN * FF;
  _Float16* Bh = (_Float16*)(mb + NN * FF);

  init_kernel<<<(NN * FF + 255) / 256, 256, 0, stream>>>(x, hb[0], xb[0]);

  for (int d = 0; d < DD; ++d) {
    const float* ew1d = ew1 + d * (2 * FF + 1) * FF;
    gemm_ab_kernel<<<64, 256, 0, stream>>>(hb[d & 1], ew1d, eb1 + d * FF, Ab, Bh);
    prep_edge_kernel<<<(NN * FF + 255) / 256, 256, 0, stream>>>(
        xb[d & 1], xb[(d + 1) & 1], mb);
    edge_kernel<<<NN / 4, 256, 0, stream>>>(
        xb[d & 1], Ab, Bh, ew1d + 2 * FF * FF /* r row of W1 */,
        eb2 + d * FF, ew2 + d * FF * FF, cw1 + d * FF * FF, cb1 + d * FF,
        cw2 + d * FF, cb2 + d, mb, xb[(d + 1) & 1]);
    node_kernel<<<NN / 128, 256, 0, stream>>>(
        hb[d & 1], mb, nw1 + d * 2 * FF * FF, nb1 + d * FF,
        nw2 + d * FF * FF, nb2 + d * FF, hb[(d + 1) & 1]);
  }
  final_kernel<<<(NN + 255) / 256, 256, 0, stream>>>(hb[DD & 1], xb[DD & 1], fw, out);
}
